// EncoderBlock_77206332113688
// MI455X (gfx1250) — matrix operand order repaired and checked by reading, not compiled, not run on hardware
//
#include <hip/hip_runtime.h>
#include <math.h>

typedef __bf16 bf16_t;
typedef __attribute__((ext_vector_type(16))) __bf16 v16bf;
typedef __attribute__((ext_vector_type(8)))  __bf16 v8bf;
typedef __attribute__((ext_vector_type(8)))  float  v8f;

union FragBF {
    v16bf v;
    v8bf  h[2];
};

#define WMMA_BF16(a, b, c) \
    __builtin_amdgcn_wmma_f32_16x16x32_bf16(false, (a), false, (b), (short)0, (c), false, false)

static constexpr int Bb  = 4;
static constexpr int Pp  = 2048;
static constexpr int Dd  = 768;
static constexpr int Hh  = 12;
static constexpr int HD  = 64;      // head dim
static constexpr int DFFc = 3072;

// ---------------------------------------------------------------------------
// DPP16 butterfly reductions across a 16-lane row (wave32 has 2 DPP rows,
// matching the two row-halves of the WMMA C layout). Replaces ds_bpermute
// shuffles with VALU ops that co-execute with the WMMA pipe.
//   0xB1  = quad_perm(1,0,3,2)  -> xor 1
//   0x4E  = quad_perm(2,3,0,1)  -> xor 2
//   0x141 = ROW_HALF_MIRROR     -> pairs quads across 8-lane halves
//   0x140 = ROW_MIRROR          -> pairs 8-lane halves across the 16-row
// ---------------------------------------------------------------------------
template <int CTRL>
__device__ __forceinline__ float dpp_movf(float v) {
    int r = __builtin_amdgcn_update_dpp(0, __float_as_int(v), CTRL, 0xF, 0xF, true);
    return __int_as_float(r);
}
__device__ __forceinline__ float rowmax16(float v) {
    v = fmaxf(v, dpp_movf<0xB1>(v));
    v = fmaxf(v, dpp_movf<0x4E>(v));
    v = fmaxf(v, dpp_movf<0x141>(v));
    v = fmaxf(v, dpp_movf<0x140>(v));
    return v;
}
__device__ __forceinline__ float rowsum16(float v) {
    v += dpp_movf<0xB1>(v);
    v += dpp_movf<0x4E>(v);
    v += dpp_movf<0x141>(v);
    v += dpp_movf<0x140>(v);
    return v;
}

// ---------------------------------------------------------------------------
// Weight convert + transpose: w[K][N] f32  ->  wt[N][K] bf16 (K contiguous)
// ---------------------------------------------------------------------------
__global__ void cvt_transpose(const float* __restrict__ w, bf16_t* __restrict__ wt,
                              int Kd, int Nd) {
    int idx = blockIdx.x * 256 + threadIdx.x;
    if (idx >= Kd * Nd) return;
    int k = idx / Nd;
    int n = idx - k * Nd;
    wt[(size_t)n * Kd + k] = (bf16_t)w[idx];
}

// ---------------------------------------------------------------------------
// Row LayerNorm (D=768) -> bf16 output.  One 256-thread block per row.
// ---------------------------------------------------------------------------
__global__ __launch_bounds__(256) void ln_bf16(const float* __restrict__ x,
                                               const float* __restrict__ g,
                                               const float* __restrict__ b,
                                               bf16_t* __restrict__ out) {
    __shared__ float red[256];
    const int tid = threadIdx.x;
    const float* xr = x + (size_t)blockIdx.x * Dd;
    float v0 = xr[tid], v1 = xr[tid + 256], v2 = xr[tid + 512];

    red[tid] = v0 + v1 + v2;
    __syncthreads();
    for (int st = 128; st > 0; st >>= 1) {
        if (tid < st) red[tid] += red[tid + st];
        __syncthreads();
    }
    const float mean = red[0] * (1.0f / Dd);
    __syncthreads();

    float d0 = v0 - mean, d1 = v1 - mean, d2 = v2 - mean;
    red[tid] = d0 * d0 + d1 * d1 + d2 * d2;
    __syncthreads();
    for (int st = 128; st > 0; st >>= 1) {
        if (tid < st) red[tid] += red[tid + st];
        __syncthreads();
    }
    const float var = red[0] * (1.0f / Dd);
    const float r = rsqrtf(var + 1e-3f);

    bf16_t* orow = out + (size_t)blockIdx.x * Dd;
    orow[tid]       = (bf16_t)(d0 * r * g[tid]       + b[tid]);
    orow[tid + 256] = (bf16_t)(d1 * r * g[tid + 256] + b[tid + 256]);
    orow[tid + 512] = (bf16_t)(d2 * r * g[tid + 512] + b[tid + 512]);
}

// ---------------------------------------------------------------------------
// WMMA GEMM:  C[M,N] = A[M,K](bf16) @ Bt[N,K](bf16)^T + bias
// Block = 256 threads = 8 waves, block tile 256x64.
// Wave tile 32x64: two A fragments (rows m0, m0+16) share each B fragment,
// so 12 b128 loads feed 8 WMMAs per K-step.
// EPI 0: scatter to q/k/v^T (head-major bf16)
// EPI 1: exact GELU -> bf16
// EPI 2: y*0.9 + resid -> f32 out
// ---------------------------------------------------------------------------
template <int EPI>
__global__ __launch_bounds__(256) void wmma_gemm(
    const bf16_t* __restrict__ A, const bf16_t* __restrict__ Bt,
    const float* __restrict__ bias, int K, int N,
    bf16_t* __restrict__ out_bf,
    float* __restrict__ out_f, const float* __restrict__ resid,
    bf16_t* __restrict__ qb, bf16_t* __restrict__ kb, bf16_t* __restrict__ vt) {

    const int lane = threadIdx.x & 31;
    const int wv   = threadIdx.x >> 5;
    const int l16  = lane & 15;
    const int hf   = lane >> 4;
    const int m0   = blockIdx.x * 256 + wv * 32;
    const int n0   = blockIdx.y * 64;

    v8f acc[2][4] = {};
    const bf16_t* arow0 = A + (size_t)(m0 + l16) * K;
    const bf16_t* arow1 = arow0 + (size_t)16 * K;
    const bf16_t* bbase = Bt + (size_t)(n0 + l16) * K;

    for (int k0 = 0; k0 < K; k0 += 32) {
        // speculative stream-ahead (8 K-steps): global_prefetch_b8
        __builtin_prefetch((const void*)(arow0 + k0 + 256), 0, 3);
        __builtin_prefetch((const void*)(bbase + k0 + 256), 0, 3);

        FragBF af0, af1;
        af0.h[0] = *(const v8bf*)(arow0 + k0 + hf * 8);        // K = hf*8 + 0..7
        af0.h[1] = *(const v8bf*)(arow0 + k0 + 16 + hf * 8);   // K = 16 + hf*8 + 0..7
        af1.h[0] = *(const v8bf*)(arow1 + k0 + hf * 8);
        af1.h[1] = *(const v8bf*)(arow1 + k0 + 16 + hf * 8);
#pragma unroll
        for (int t = 0; t < 4; ++t) {
            const bf16_t* bcol = bbase + (size_t)(t * 16) * K + k0 + hf * 8;
            FragBF bfr;
            bfr.h[0] = *(const v8bf*)(bcol);                   // K = hf*8 + 0..7
            bfr.h[1] = *(const v8bf*)(bcol + 16);              // K = 16 + hf*8 + 0..7
            acc[0][t] = WMMA_BF16(af0.v, bfr.v, acc[0][t]);
            acc[1][t] = WMMA_BF16(af1.v, bfr.v, acc[1][t]);
        }
    }

#pragma unroll
    for (int r = 0; r < 2; ++r) {
#pragma unroll
        for (int t = 0; t < 4; ++t) {
            const int n = n0 + t * 16 + l16;       // C layout: col = lane%16
            const float bv = bias[n];
#pragma unroll
            for (int v = 0; v < 8; ++v) {
                const int m = m0 + r * 16 + v + 8 * hf;  // row = v + 8*(lane/16)
                float val = acc[r][t][v] + bv;
                if constexpr (EPI == 0) {
                    const int bidx = m >> 11;        // / P
                    const int p    = m & (Pp - 1);
                    const int seg  = n / Dd;         // 0=q 1=k 2=v
                    const int d    = n - seg * Dd;
                    const int head = d >> 6;
                    const int c    = d & (HD - 1);
                    const size_t bh = (size_t)(bidx * Hh + head);
                    if (seg == 0)      qb[(bh * Pp + p) * HD + c] = (bf16_t)val;
                    else if (seg == 1) kb[(bh * Pp + p) * HD + c] = (bf16_t)val;
                    else               vt[(bh * HD + c) * Pp + p] = (bf16_t)val; // V^T
                } else if constexpr (EPI == 1) {
                    const float gl = 0.5f * val * (1.0f + erff(val * 0.70710678118f));
                    out_bf[(size_t)m * N + n] = (bf16_t)gl;
                } else {
                    const size_t i = (size_t)m * N + n;
                    out_f[i] = val * 0.9f + resid[i];
                }
            }
        }
    }
}

// ---------------------------------------------------------------------------
// Flash attention: grid (B*H, P/64), block 128 (4 waves, 16 q-rows each).
// Streams 32-key tiles; online softmax (DPP row reductions); O accumulated
// via WMMA against V^T.  Epilogue: x1 = x + 0.9 * softmax(QK^T/8) V
// ---------------------------------------------------------------------------
__global__ __launch_bounds__(128) void attn_flash(
    const bf16_t* __restrict__ qb, const bf16_t* __restrict__ kb,
    const bf16_t* __restrict__ vt, const float* __restrict__ x,
    float* __restrict__ x1) {

    __shared__ alignas(16) bf16_t probs[4][16][32];

    const int lane = threadIdx.x & 31;
    const int wv   = threadIdx.x >> 5;
    const int l16  = lane & 15;
    const int hf   = lane >> 4;
    const int bh   = blockIdx.x;
    const int q0   = blockIdx.y * 64 + wv * 16;

    // Preload Q fragments for this wave's 16 rows (hd=64 -> two K=32 steps)
    const bf16_t* qrow = qb + ((size_t)bh * Pp + q0 + l16) * HD;
    FragBF qf[2];
#pragma unroll
    for (int ks = 0; ks < 2; ++ks) {
        qf[ks].h[0] = *(const v8bf*)(qrow + ks * 32 + hf * 8);
        qf[ks].h[1] = *(const v8bf*)(qrow + ks * 32 + 16 + hf * 8);
    }

    v8f o[4] = {};
    float mrow[8], lrow[8];
#pragma unroll
    for (int v = 0; v < 8; ++v) { mrow[v] = -3.0e38f; lrow[v] = 0.0f; }

    const bf16_t* kbh = kb + (size_t)bh * Pp * HD;
    const bf16_t* vbh = vt + (size_t)bh * HD * Pp;

    for (int kt = 0; kt < Pp; kt += 32) {
        // stream-ahead for next key tile
        __builtin_prefetch((const void*)(kbh + (size_t)(kt + 32 + l16) * HD), 0, 3);
        __builtin_prefetch((const void*)(vbh + (size_t)l16 * Pp + kt + 32), 0, 3);

        // S = Q @ K^T for 16 queries x 32 keys (two 16x16 tiles)
        v8f s[2] = {};
#pragma unroll
        for (int nt = 0; nt < 2; ++nt) {
#pragma unroll
            for (int ks = 0; ks < 2; ++ks) {
                const bf16_t* kp = kbh + (size_t)(kt + nt * 16 + l16) * HD
                                       + ks * 32 + hf * 8;
                FragBF kf;
                kf.h[0] = *(const v8bf*)(kp);
                kf.h[1] = *(const v8bf*)(kp + 16);
                s[nt] = WMMA_BF16(qf[ks].v, kf.v, s[nt]);
            }
        }
        // Online softmax update. Row r = v + 8*hf is replicated across a
        // 16-lane DPP row, so the DPP butterfly is an exact row reduction.
#pragma unroll
        for (int v = 0; v < 8; ++v) {
            float s0 = s[0][v] * 0.125f;     // / sqrt(hd)
            float s1 = s[1][v] * 0.125f;
            float mx = rowmax16(fmaxf(s0, s1));
            const float mn = fmaxf(mrow[v], mx);
            const float alpha = __expf(mrow[v] - mn);
            const float p0 = __expf(s0 - mn);
            const float p1 = __expf(s1 - mn);
            const float ps = rowsum16(p0 + p1);
            lrow[v] = lrow[v] * alpha + ps;
            mrow[v] = mn;
#pragma unroll
            for (int t = 0; t < 4; ++t) o[t][v] *= alpha;
            // C-layout -> LDS (row-major 16x32 probs, bf16)
            probs[wv][v + 8 * hf][l16]      = (bf16_t)p0;
            probs[wv][v + 8 * hf][16 + l16] = (bf16_t)p1;
        }
        // Reload probs in A-fragment layout (wave-private LDS region, no barrier)
        FragBF pf;
        const bf16_t* pr = &probs[wv][l16][0];
        pf.h[0] = *(const v8bf*)(pr + hf * 8);
        pf.h[1] = *(const v8bf*)(pr + 16 + hf * 8);
        // O += P @ V   (V stored transposed [hd][P] -> K-contiguous b128 loads)
#pragma unroll
        for (int ct = 0; ct < 4; ++ct) {
            const bf16_t* vp = vbh + (size_t)(ct * 16 + l16) * Pp + kt + hf * 8;
            FragBF vf;
            vf.h[0] = *(const v8bf*)(vp);
            vf.h[1] = *(const v8bf*)(vp + 16);
            o[ct] = WMMA_BF16(pf.v, vf.v, o[ct]);
        }
    }

    // Epilogue: normalize, droppath-scale, residual, scatter back to [B,P,D]
    const int bidx = bh / Hh;
    const int head = bh - bidx * Hh;
#pragma unroll
    for (int ct = 0; ct < 4; ++ct) {
        const int d = head * HD + ct * 16 + l16;
#pragma unroll
        for (int v = 0; v < 8; ++v) {
            const int p = q0 + v + 8 * hf;
            const size_t i = ((size_t)bidx * Pp + p) * Dd + d;
            x1[i] = x[i] + 0.9f * (o[ct][v] / lrow[v]);
        }
    }
}

// ---------------------------------------------------------------------------
extern "C" void kernel_launch(void* const* d_in, const int* in_sizes, int n_in,
                              void* d_out, int out_size, void* d_ws, size_t ws_size,
                              hipStream_t stream) {
    (void)in_sizes; (void)n_in; (void)out_size; (void)ws_size;
    const float* x     = (const float*)d_in[0];
    const float* ln1_g = (const float*)d_in[1];
    const float* ln1_b = (const float*)d_in[2];
    const float* w_qkv = (const float*)d_in[3];
    const float* b_qkv = (const float*)d_in[4];
    const float* ln2_g = (const float*)d_in[5];
    const float* ln2_b = (const float*)d_in[6];
    const float* w1    = (const float*)d_in[7];
    const float* b1    = (const float*)d_in[8];
    const float* w2    = (const float*)d_in[9];
    const float* b2    = (const float*)d_in[10];
    float* out = (float*)d_out;

    char* ws = (char*)d_ws;
    size_t off = 0;
    auto take = [&](size_t bytes) -> char* {
        char* p = ws + off;
        off = (off + bytes + 255) & ~(size_t)255;
        return p;
    };
    const size_t M = (size_t)Bb * Pp;  // 8192 rows

    bf16_t* wqkv_t = (bf16_t*)take((size_t)3 * Dd * Dd * 2);   // [3D][D]
    bf16_t* w1_t   = (bf16_t*)take((size_t)Dd * DFFc * 2);     // [DFF][D]
    bf16_t* w2_t   = (bf16_t*)take((size_t)Dd * DFFc * 2);     // [D][DFF]
    bf16_t* hbf    = (bf16_t*)take(M * Dd * 2);                // LN1(x) bf16
    bf16_t* qb_    = (bf16_t*)take(M * Dd * 2);                // [B,H,P,hd]
    bf16_t* kb_    = (bf16_t*)take(M * Dd * 2);                // [B,H,P,hd]
    bf16_t* vt_    = (bf16_t*)take(M * Dd * 2);                // [B,H,hd,P]
    float*  x1     = (float*)take(M * Dd * 4);                 // attn residual
    bf16_t* h2bf   = (bf16_t*)take(M * Dd * 2);                // LN2(x1) bf16
    bf16_t* gbf    = (bf16_t*)take(M * DFFc * 2);              // gelu(h@w1+b1)

    // Weight downconvert + transpose (K-contiguous for WMMA B fragments)
    { int tot = Dd * 3 * Dd; cvt_transpose<<<(tot + 255) / 256, 256, 0, stream>>>(w_qkv, wqkv_t, Dd, 3 * Dd); }
    { int tot = Dd * DFFc;   cvt_transpose<<<(tot + 255) / 256, 256, 0, stream>>>(w1, w1_t, Dd, DFFc); }
    { int tot = DFFc * Dd;   cvt_transpose<<<(tot + 255) / 256, 256, 0, stream>>>(w2, w2_t, DFFc, Dd); }

    // LN1
    ln_bf16<<<dim3((unsigned)M), 256, 0, stream>>>(x, ln1_g, ln1_b, hbf);

    // QKV projection (scatter to head-major q/k/v^T)
    {
        dim3 g((unsigned)(M / 256), (3 * Dd) / 64);
        wmma_gemm<0><<<g, 256, 0, stream>>>(hbf, wqkv_t, b_qkv, Dd, 3 * Dd,
                                            nullptr, nullptr, nullptr, qb_, kb_, vt_);
    }

    // Flash attention + droppath + residual -> x1
    {
        dim3 g(Bb * Hh, Pp / 64);
        attn_flash<<<g, 128, 0, stream>>>(qb_, kb_, vt_, x, x1);
    }

    // LN2
    ln_bf16<<<dim3((unsigned)M), 256, 0, stream>>>(x1, ln2_g, ln2_b, h2bf);

    // MLP up-projection + exact GELU -> bf16
    {
        dim3 g((unsigned)(M / 256), DFFc / 64);
        wmma_gemm<1><<<g, 256, 0, stream>>>(h2bf, w1_t, b1, Dd, DFFc,
                                            gbf, nullptr, nullptr, nullptr, nullptr, nullptr);
    }

    // MLP down-projection + droppath + residual -> out
    {
        dim3 g((unsigned)(M / 256), Dd / 64);
        wmma_gemm<2><<<g, 256, 0, stream>>>(gbf, w2_t, b2, DFFc, Dd,
                                            nullptr, out, x1, nullptr, nullptr, nullptr);
    }
}
